// ValleFlashAttention_63178968924361
// MI455X (gfx1250) — compile-verified, matrix operand order checked by reading, not run on hardware
//
#include <hip/hip_runtime.h>
#include <hip/hip_bf16.h>

// ---------------------------------------------------------------------------
// MI455X (gfx1250) attention block.
// B=2, N=2048, C=1024, H=16, hd=64, causal_start=16, tail-128 padding.
// Compute-bound (~70 GFLOP vs ~100MB traffic @23.3TB/s) -> bf16 WMMA
// (v_wmma_f32_16x16x32_bf16) with f32 accumulation everywhere.
// ---------------------------------------------------------------------------

typedef __bf16 bf16;
typedef __attribute__((ext_vector_type(16))) __bf16 v16bf;
typedef __attribute__((ext_vector_type(8)))  float  v8f;

#define WMMA_BF16(a, b, c) \
  __builtin_amdgcn_wmma_f32_16x16x32_bf16(false, (a), false, (b), (short)0, (c), false, false)

__device__ __forceinline__ bf16 f2bf(float f) {
  unsigned u = __builtin_bit_cast(unsigned, f);
  u += 0x7FFFu + ((u >> 16) & 1u);               // round-to-nearest-even
  unsigned short h = (unsigned short)(u >> 16);
  return __builtin_bit_cast(bf16, h);
}

// Load one 16-lane-striped WMMA operand fragment (A: row=m, B: row=n' of W /
// column of B). ISA layout: lanes 0-15 hold K = [k0,k0+8) U [k0+16,k0+24),
// lanes 16-31 hold K = [k0+8,k0+16) U [k0+24,k0+32).  Two 16B loads per lane.
__device__ __forceinline__ v16bf load_frag(const bf16* __restrict__ base,
                                           int row, int ld, int k0, int half) {
  const bf16* p = base + (size_t)row * (size_t)ld + (size_t)(k0 + half * 8);
  uint4 lo = *(const uint4*)(p);
  uint4 hi = *(const uint4*)(p + 16);
  v16bf r;
  ((uint4*)&r)[0] = lo;
  ((uint4*)&r)[1] = hi;
  return r;
}

// ---------------------------------------------------------------------------
// fp32 -> bf16 conversion pass
// ---------------------------------------------------------------------------
__global__ void cvt_f32_bf16(const float* __restrict__ in, bf16* __restrict__ out, int n) {
  int i = blockIdx.x * blockDim.x + threadIdx.x;
  if (i < n) out[i] = f2bf(in[i]);
}

// ---------------------------------------------------------------------------
// QKV projection: qkv = x(4096x1024) @ w_qkv^T (1024x3072).
// 8 waves/block, each wave computes a 16x64 tile (4 accumulators, A reused 4x).
// Epilogue: q -> qb (bf16, [bh][n][d]); k -> kb (bf16) + present[0] (f32);
//           v -> vt (bf16 TRANSPOSED [bh][d][n]) + present[1] (f32).
// ---------------------------------------------------------------------------
__global__ void __launch_bounds__(256)
qkv_gemm(const bf16* __restrict__ xb, const bf16* __restrict__ wb,
         bf16* __restrict__ qb, bf16* __restrict__ kb, bf16* __restrict__ vt,
         float* __restrict__ present, int B_, int N_, int C_) {
  const int H = 16, HD = 64;
  const int wave = threadIdx.x >> 5;
  const int lane = threadIdx.x & 31;
  const int lh = lane & 15, half = lane >> 4;
  const int row0 = blockIdx.x * 128 + wave * 16;   // M tile base
  const int col0 = blockIdx.y * 64;                // N' tile base (within 3C)

  v8f acc[4] = {};
  for (int k0 = 0; k0 < C_; k0 += 32) {
    v16bf a = load_frag(xb, row0 + lh, C_, k0, half);
#pragma unroll
    for (int t = 0; t < 4; ++t) {
      v16bf b = load_frag(wb, col0 + t * 16 + lh, C_, k0, half);
      acc[t] = WMMA_BF16(a, b, acc[t]);
    }
  }

  const long NH = (long)N_ * HD;
#pragma unroll
  for (int t = 0; t < 4; ++t) {
    int c = col0 + t * 16 + lh;     // this lane's output column in [0,3C)
    int s = c / C_;                 // 0:q 1:k 2:v
    int rem = c % C_;
    int h = rem / HD, d = rem % HD;
#pragma unroll
    for (int i = 0; i < 8; ++i) {
      int r = row0 + i + 8 * half;  // D layout: row = elem + 8*(lane/16)
      int b_ = r / N_, n = r % N_;
      float v = acc[t][i];
      long bh = (long)(b_ * H + h);
      long idx = bh * NH + (long)n * HD + d;
      if (s == 0) {
        qb[idx] = f2bf(v);
      } else if (s == 1) {
        kb[idx] = f2bf(v);
        present[idx] = v;                                   // present[0,b,h,n,d]
      } else {
        vt[bh * NH + (long)d * N_ + n] = f2bf(v);           // transposed for P@V
        present[(long)B_ * H * NH + idx] = v;               // present[1,b,h,n,d]
      }
    }
  }
}

// ---------------------------------------------------------------------------
// Flash attention. Grid: (B*H, N/128). 8 waves/block; each wave owns a
// 16-row Q tile. Keys processed in chunks of 32 (2 score tiles -> one P
// A-fragment). Online softmax; row stats reduced with shfl_xor across the
// 16-lane groups of the C/D layout. P transposed to A layout through LDS.
// ---------------------------------------------------------------------------
__global__ void __launch_bounds__(256)
flash_attn(const bf16* __restrict__ qb, const bf16* __restrict__ kb,
           const bf16* __restrict__ vt, const unsigned char* __restrict__ pad,
           const int* __restrict__ causal_start_p,
           bf16* __restrict__ ob, int B_, int N_) {
  const int H = 16, HD = 64;
  __shared__ __align__(16) bf16 pbuf[8][16][32];   // per-wave P staging (8KB)

  const int wave = threadIdx.x >> 5;
  const int lane = threadIdx.x & 31;
  const int lh = lane & 15, half = lane >> 4;
  const int bh = blockIdx.x;
  const int b_ = bh / H, h = bh % H;
  const int q0 = blockIdx.y * 128 + wave * 16;
  const int cstart = causal_start_p[0];

  const bf16* qp = qb + (long)bh * N_ * HD;
  const bf16* kp = kb + (long)bh * N_ * HD;
  const bf16* vp = vt + (long)bh * HD * N_;
  const unsigned char* padb = pad + (long)b_ * N_;

  // Q fragments (hd chunks 0 and 32) — loaded once, reused every key chunk.
  v16bf qa0 = load_frag(qp, q0 + lh, HD, 0, half);
  v16bf qa1 = load_frag(qp, q0 + lh, HD, 32, half);

  v8f o[4] = {};
  float mrow[8], lrow[8];
#pragma unroll
  for (int i = 0; i < 8; ++i) { mrow[i] = -3.0e38f; lrow[i] = 0.0f; }
  const float scale = 0.125f;   // hd^-0.5, hd = 64

  // Causal skip: keys allowed iff k <= q or k < cstart.
  int kend = q0 + 16;
  if (cstart > kend) kend = cstart;
  kend = (kend + 31) & ~31;
  if (kend > N_) kend = N_;

  for (int kc = 0; kc < kend; kc += 32) {
    // --- S = Q K^T for 32 keys (two 16-col tiles, hd accumulated over 2 WMMAs)
    v8f s0 = {}, s1 = {};
    {
      v16bf b0 = load_frag(kp, kc + lh, HD, 0, half);
      v16bf b1 = load_frag(kp, kc + lh, HD, 32, half);
      s0 = WMMA_BF16(qa0, b0, s0);
      s0 = WMMA_BF16(qa1, b1, s0);
      v16bf b2 = load_frag(kp, kc + 16 + lh, HD, 0, half);
      v16bf b3 = load_frag(kp, kc + 16 + lh, HD, 32, half);
      s1 = WMMA_BF16(qa0, b2, s1);
      s1 = WMMA_BF16(qa1, b3, s1);
    }

    // --- mask + scale
    const int k0a = kc + lh, k1a = kc + 16 + lh;
    const bool ok0 = !padb[k0a], ok1 = !padb[k1a];
    float tmax[8];
#pragma unroll
    for (int i = 0; i < 8; ++i) {
      int qa = q0 + i + 8 * half;
      float v0 = (ok0 && (k0a <= qa || k0a < cstart)) ? s0[i] * scale : -3.0e38f;
      float v1 = (ok1 && (k1a <= qa || k1a < cstart)) ? s1[i] * scale : -3.0e38f;
      s0[i] = v0; s1[i] = v1;
      tmax[i] = fmaxf(v0, v1);
    }
    // --- row max across the 16-lane group (xor offsets 1..8 stay in-group)
#pragma unroll
    for (int off = 1; off < 16; off <<= 1)
#pragma unroll
      for (int i = 0; i < 8; ++i)
        tmax[i] = fmaxf(tmax[i], __shfl_xor(tmax[i], off, 32));

    // --- online softmax update
    float cf[8], rsum[8];
#pragma unroll
    for (int i = 0; i < 8; ++i) {
      float mnew = fmaxf(mrow[i], tmax[i]);
      cf[i] = __expf(mrow[i] - mnew);
      mrow[i] = mnew;
      float p0 = __expf(s0[i] - mnew);
      float p1 = __expf(s1[i] - mnew);
      s0[i] = p0; s1[i] = p1;
      rsum[i] = p0 + p1;
    }
#pragma unroll
    for (int off = 1; off < 16; off <<= 1)
#pragma unroll
      for (int i = 0; i < 8; ++i)
        rsum[i] += __shfl_xor(rsum[i], off, 32);
#pragma unroll
    for (int i = 0; i < 8; ++i) lrow[i] = lrow[i] * cf[i] + rsum[i];
#pragma unroll
    for (int t = 0; t < 4; ++t)
#pragma unroll
      for (int i = 0; i < 8; ++i)
        o[t][i] *= cf[i];

    // --- transpose P (D layout) to A layout via per-wave LDS tile
#pragma unroll
    for (int i = 0; i < 8; ++i) {
      int m = i + 8 * half;
      pbuf[wave][m][lh]      = f2bf(s0[i]);
      pbuf[wave][m][16 + lh] = f2bf(s1[i]);
    }
    asm volatile("s_wait_dscnt 0" ::: "memory");   // wave-local LDS RAW fence
    v16bf pa = load_frag(&pbuf[wave][0][0], lh, 32, 0, half);

    // --- O += P @ V  (V^T is [hd][key], contiguous keys per lane)
#pragma unroll
    for (int t = 0; t < 4; ++t) {
      v16bf vb = load_frag(vp, t * 16 + lh, N_, kc, half);
      o[t] = WMMA_BF16(pa, vb, o[t]);
    }
  }

  // --- normalize and store attention output as bf16 (B,N,C) row-major
#pragma unroll
  for (int t = 0; t < 4; ++t) {
    int d = t * 16 + lh;
#pragma unroll
    for (int i = 0; i < 8; ++i) {
      int qa = q0 + i + 8 * half;
      float val = o[t][i] / lrow[i];
      ob[((long)(b_ * N_ + qa)) * (H * HD) + h * HD + d] = f2bf(val);
    }
  }
}

// ---------------------------------------------------------------------------
// Output projection: out = attn(4096x1024) @ w_proj^T + b_proj  (fp32 out)
// ---------------------------------------------------------------------------
__global__ void __launch_bounds__(256)
proj_gemm(const bf16* __restrict__ ob, const bf16* __restrict__ wb,
          const float* __restrict__ bias, float* __restrict__ out, int C_) {
  const int wave = threadIdx.x >> 5;
  const int lane = threadIdx.x & 31;
  const int lh = lane & 15, half = lane >> 4;
  const int row0 = blockIdx.x * 128 + wave * 16;
  const int col0 = blockIdx.y * 64;

  v8f acc[4] = {};
  for (int k0 = 0; k0 < C_; k0 += 32) {
    v16bf a = load_frag(ob, row0 + lh, C_, k0, half);
#pragma unroll
    for (int t = 0; t < 4; ++t) {
      v16bf b = load_frag(wb, col0 + t * 16 + lh, C_, k0, half);
      acc[t] = WMMA_BF16(a, b, acc[t]);
    }
  }
#pragma unroll
  for (int t = 0; t < 4; ++t) {
    int c = col0 + t * 16 + lh;
    float bv = bias[c];
#pragma unroll
    for (int i = 0; i < 8; ++i) {
      int r = row0 + i + 8 * half;
      out[(long)r * C_ + c] = acc[t][i] + bv;
    }
  }
}

// ---------------------------------------------------------------------------
// Host launcher
// ---------------------------------------------------------------------------
extern "C" void kernel_launch(void* const* d_in, const int* in_sizes, int n_in,
                              void* d_out, int out_size, void* d_ws, size_t ws_size,
                              hipStream_t stream) {
  (void)in_sizes; (void)n_in; (void)out_size; (void)ws_size;
  const int B = 2, N = 2048, C = 1024, H = 16, HD = 64;

  const float*         x      = (const float*)d_in[0];
  const unsigned char* pad    = (const unsigned char*)d_in[1]; // jnp bool -> 1 byte
  const int*           cstart = (const int*)d_in[2];
  const float*         w_qkv  = (const float*)d_in[3];
  const float*         w_proj = (const float*)d_in[4];
  const float*         b_proj = (const float*)d_in[5];

  float* out     = (float*)d_out;
  float* present = out + (size_t)B * N * C;   // (2,B,H,N,hd) fp32 after `out`

  // Workspace carve-up (~48 MB, 256B aligned)
  char* ws = (char*)d_ws;
  size_t off = 0;
  auto carve = [&](size_t bytes) -> void* {
    void* p = ws + off;
    off = (off + bytes + 255) & ~(size_t)255;
    return p;
  };
  bf16* xb     = (bf16*)carve((size_t)B * N * C * 2);
  bf16* wqkvb  = (bf16*)carve((size_t)3 * C * C * 2);
  bf16* wprojb = (bf16*)carve((size_t)C * C * 2);
  bf16* qb     = (bf16*)carve((size_t)B * H * N * HD * 2);
  bf16* kb     = (bf16*)carve((size_t)B * H * N * HD * 2);
  bf16* vt     = (bf16*)carve((size_t)B * H * HD * N * 2);
  bf16* ob     = (bf16*)carve((size_t)B * N * C * 2);

  const int n1 = B * N * C, n2 = 3 * C * C, n3 = C * C;
  cvt_f32_bf16<<<(n1 + 255) / 256, 256, 0, stream>>>(x, xb, n1);
  cvt_f32_bf16<<<(n2 + 255) / 256, 256, 0, stream>>>(w_qkv, wqkvb, n2);
  cvt_f32_bf16<<<(n3 + 255) / 256, 256, 0, stream>>>(w_proj, wprojb, n3);

  dim3 g1(B * N / 128, 3 * C / 64);
  qkv_gemm<<<g1, 256, 0, stream>>>(xb, wqkvb, qb, kb, vt, present, B, N, C);

  dim3 g2(B * H, N / 128);
  flash_attn<<<g2, 256, 0, stream>>>(qb, kb, vt, pad, cstart, ob, B, N);

  dim3 g3(B * N / 128, C / 64);
  proj_gemm<<<g3, 256, 0, stream>>>(ob, wprojb, b_proj, out, C);
}
